// RSFconv_18588618457374
// MI455X (gfx1250) — matrix operand order repaired and checked by reading, compiled, not run on hardware
//
#include <hip/hip_runtime.h>
#include <hip/hip_bf16.h>
#include <cstdint>

typedef __attribute__((ext_vector_type(16))) _Float16 v16h;
typedef __attribute__((ext_vector_type(8)))  _Float16 v8h;
typedef __attribute__((ext_vector_type(8)))  float    v8f;

#define HW      128
#define BATCH   8
#define CIN     256      // IN_NUM * ROT_NUM * SCALE_NUM
#define MOUT    128      // OUT_NUM * ROT_NUM (per scale)
#define KBASIS  30       // 2 * INP * ceil(INP/2)

// ---- CDNA5 async global->LDS path (guarded; falls back to sync copy) ------
#if defined(__has_builtin)
#  if __has_builtin(__builtin_amdgcn_global_load_async_to_lds_b32)
#    define HAVE_ASYNC 1
#  endif
#endif
#ifndef HAVE_ASYNC
#  define HAVE_ASYNC 0
#endif

typedef __attribute__((address_space(1))) int as1_int;
typedef __attribute__((address_space(3))) int as3_int;

__device__ __forceinline__ void async_cp_f32(const float* g, float* l) {
#if HAVE_ASYNC
    // per-lane global src -> per-lane LDS dst (transpose comes for free).
    // generic->AS1 keeps the 64-bit value; generic shared low 32 bits == LDS addr.
    __builtin_amdgcn_global_load_async_to_lds_b32(
        (as1_int*)(unsigned long long)(uintptr_t)g,
        (as3_int*)(unsigned int)(uintptr_t)l,
        0, 0);
#else
    *l = *g;
#endif
}

__device__ __forceinline__ void wait_async() {
#if HAVE_ASYNC
#  if __has_builtin(__builtin_amdgcn_s_wait_asynccnt)
    __builtin_amdgcn_s_wait_asynccnt(0);
#  else
    asm volatile("s_wait_asynccnt 0x0" ::: "memory");
#  endif
#endif
}

// ---------------------------------------------------------------------------
// Filter synthesis: filt_s[mo=(m_out*4+o)][c=((n*2+sig')*4+a)][dy][dx]
//   = sum_k basis_s[dy][dx][o][k] * W[m_out][((sig'-s)%2)*32+n][(a-o)%4][k]
// Written pre-swizzled for the WMMA 16-bit A-fragment:
//   wsA[ ((pos*8 + cc)*128 + m) * 32 + kk ]
// ---------------------------------------------------------------------------
__global__ void prep_filters(const float* __restrict__ W,      // [32,64,4,30]
                             const float* __restrict__ basis,  // [ks,ks,4,30]
                             _Float16* __restrict__ wsA,
                             int ks, int s, int total) {
    int idx = blockIdx.x * blockDim.x + threadIdx.x;
    if (idx >= total) return;
    int kk  = idx & 31;
    int m   = (idx >> 5) & 127;
    int cc  = (idx >> 12) & 7;
    int pos = idx >> 15;               // dy*ks + dx
    int c   = cc * 32 + kk;            // input channel
    int a   = c & 3;
    int sig = (c >> 2) & 1;
    int n   = c >> 3;
    int m_out = m >> 2;
    int o     = m & 3;
    int sig_src = (sig - s) & 1;       // scale cyclic roll by s
    int a_src   = (a - o) & 3;         // rotation cyclic permutation

    const float* bp = basis + (pos * 4 + o) * KBASIS;
    const float* wp = W + (((m_out * 64) + sig_src * 32 + n) * 4 + a_src) * KBASIS;
    float acc = 0.f;
#pragma unroll
    for (int k = 0; k < KBASIS; ++k) acc += bp[k] * wp[k];
    wsA[idx] = (_Float16)acc;
}

// ---------------------------------------------------------------------------
// Implicit-GEMM conv, double-buffered async pipeline:
//   phase(cc,dy): ASYNC copy f32 row tile -> sF[next] while WMMAs run on sH;
//   per phase: wait_asynccnt -> barrier -> issue next async -> cvt f32->f16
//   into sH -> barrier -> ks shifted v_wmma_f32_16x16x32_f16 steps.
// Block = 8 wave32s, 128(M) x 128(one image row). Wave: 64x32 acc (4x2 frags).
// ---------------------------------------------------------------------------
template <int KS>
__global__ __launch_bounds__(256) void conv_wmma(const float* __restrict__ x,
                                                 const _Float16* __restrict__ wsA,
                                                 const float* __restrict__ cbias,
                                                 float* __restrict__ out,
                                                 int chan_off) {
    constexpr int PAD  = (KS - 1) / 2;
    constexpr int COLS = HW + 2 * PAD;
    constexpr int NPH  = 8 * KS;                 // (cin/32) * KS phases
    __shared__ float    sF[2][COLS * 32];        // f32 staging, [col][32ch]
    __shared__ _Float16 sH[COLS * 32];           // f16 compute tile

    const int t     = threadIdx.x;
    const int lane  = t & 31;
    const int wv    = t >> 5;
    const int waveM = wv >> 2;        // 0..1 -> M half (64 rows)
    const int waveN = wv & 3;         // 0..3 -> 32-pixel strip
    const int lhalf = lane >> 4;      // 0/1
    const int lm    = lane & 15;

    const int n_img = blockIdx.x >> 7;           // 8 images * 128 rows
    const int y     = blockIdx.x & 127;

    // zero the halo columns of both f32 buffers once (async never writes them)
    for (int idx = t; idx < 2 * 2 * PAD * 32; idx += 256) {
        int b    = idx / (2 * PAD * 32);
        int rem  = idx - b * (2 * PAD * 32);
        int side = rem / (PAD * 32);
        int rem2 = rem - side * (PAD * 32);
        int p    = rem2 >> 5;
        int ch   = rem2 & 31;
        int col  = side ? (HW + PAD + p) : p;
        sF[b][col * 32 + ch] = 0.f;
    }

    // stage phase ph (cc = ph/KS, dy = ph%KS) into sF[buf]
    auto stage = [&](int ph, int buf) {
        const int  cc    = ph / KS;
        const int  dy    = ph - cc * KS;
        const int  row   = y + dy - PAD;
        const bool rowok = (row >= 0) && (row < HW);
#pragma unroll
        for (int j = 0; j < 4; ++j) {
            const int ch = (t >> 5) + 8 * j;     // 0..31
            const float* src = x + (((size_t)n_img * CIN + cc * 32 + ch) * HW + (rowok ? row : 0)) * (size_t)HW;
#pragma unroll
            for (int i = 0; i < 4; ++i) {
                const int xc = (t & 31) + 32 * i;       // 0..127, coalesced per wave
                float* dst = &sF[buf][(xc + PAD) * 32 + ch];
                if (rowok) async_cp_f32(src + xc, dst);
                else       *dst = 0.f;
            }
        }
    };

    v8f acc[4][2] = {};

    __syncthreads();       // halo zeros visible
    stage(0, 0);

    for (int ph = 0; ph < NPH; ++ph) {
        const int cur = ph & 1;
        wait_async();
        __syncthreads();                         // sF[cur] ready; sH free
        if (ph + 1 < NPH) stage(ph + 1, cur ^ 1);
        // convert current f32 tile -> f16 compute tile
        for (int idx = t; idx < COLS * 32; idx += 256)
            sH[idx] = (_Float16)sF[cur][idx];
        __syncthreads();                         // sH ready

        const int cc = ph / KS;
        const int dy = ph - cc * KS;
        if (lane == 0)
            __builtin_prefetch(wsA + ((size_t)((dy * KS) * 8 + ((cc + 1) & 7)) * 128) * 32, 0, 0);
#pragma unroll
        for (int dx = 0; dx < KS; ++dx) {
            const int pos = dy * KS + dx;
            const _Float16* Abase = wsA + ((size_t)(pos * 8 + cc) * 128) * 32;
            v16h afrag[4];
#pragma unroll
            for (int mf = 0; mf < 4; ++mf) {
                // A 16x32 f16, k order chosen here (B uses the same): lanes<16 K{0..15}, lanes>=16 K{16..31}
                const _Float16* ap = Abase + (waveM * 64 + mf * 16 + lm) * 32;
                v8h a0 = *(const v8h*)(ap + 16 * lhalf);
                v8h a1 = *(const v8h*)(ap + 16 * lhalf + 8);
                afrag[mf] = __builtin_shufflevector(a0, a1,
                    0, 1, 2, 3, 4, 5, 6, 7, 8, 9, 10, 11, 12, 13, 14, 15);
            }
            v16h bfrag[2];
#pragma unroll
            for (int nf = 0; nf < 2; ++nf) {
                // B 32x16: column = lane%16, K range = 16*(lane/16)
                const int colIdx = waveN * 32 + nf * 16 + lm + dx;
                bfrag[nf] = *(const v16h*)(&sH[colIdx * 32 + 16 * lhalf]);
            }
#pragma unroll
            for (int mf = 0; mf < 4; ++mf)
#pragma unroll
                for (int nf = 0; nf < 2; ++nf)
                    acc[mf][nf] = __builtin_amdgcn_wmma_f32_16x16x32_f16(
                        false, afrag[mf], false, bfrag[nf],
                        (short)0, acc[mf][nf], false, false);
        }
    }

    // ---- epilogue: C/D layout M = r + 8*(lane/16), N = lane%16 ----
#pragma unroll
    for (int mf = 0; mf < 4; ++mf) {
#pragma unroll
        for (int r = 0; r < 8; ++r) {
            const int    mo      = waveM * 64 + mf * 16 + 8 * lhalf + r;
            const float  b       = cbias[mo >> 2];     // bias repeated over ROT
            const size_t rowbase = (((size_t)n_img * (2 * MOUT) + chan_off + mo) * HW + y) * (size_t)HW;
#pragma unroll
            for (int nf = 0; nf < 2; ++nf) {
                const int pix = waveN * 32 + nf * 16 + lm;
                out[rowbase + pix] = acc[mf][nf][r] + b;
            }
        }
    }
}

extern "C" void kernel_launch(void* const* d_in, const int* in_sizes, int n_in,
                              void* d_out, int out_size, void* d_ws, size_t ws_size,
                              hipStream_t stream) {
    (void)in_sizes; (void)n_in; (void)out_size; (void)ws_size;
    const float* x  = (const float*)d_in[0];
    const float* W  = (const float*)d_in[1];
    const float* cb = (const float*)d_in[2];
    const float* b0 = (const float*)d_in[3];
    const float* b1 = (const float*)d_in[4];
    float* out = (float*)d_out;

    _Float16* wsA0 = (_Float16*)d_ws;                 // 9*8*128*32 halves
    _Float16* wsA1 = wsA0 + 9 * 8 * 128 * 32;         // 25*8*128*32 halves

    {
        int total = 9 * 8 * 128 * 32;
        prep_filters<<<(total + 255) / 256, 256, 0, stream>>>(W, b0, wsA0, 3, 0, total);
    }
    {
        int total = 25 * 8 * 128 * 32;
        prep_filters<<<(total + 255) / 256, 256, 0, stream>>>(W, b1, wsA1, 5, 1, total);
    }

    conv_wmma<3><<<BATCH * HW, 256, 0, stream>>>(x, wsA0, cb, out, 0);
    conv_wmma<5><<<BATCH * HW, 256, 0, stream>>>(x, wsA1, cb, out, MOUT);
}
